// WindowAttention_47175920780033
// MI455X (gfx1250) — compile-verified, matrix operand order checked by reading, not run on hardware
//
#include <hip/hip_runtime.h>
#include <hip/hip_bf16.h>

typedef __attribute__((ext_vector_type(16))) _Float16 v16h;
typedef __attribute__((ext_vector_type(8)))  _Float16 v8h;
typedef __attribute__((ext_vector_type(8)))  float    v8f;

#define DEVI __device__ __forceinline__

constexpr int B_   = 128;
constexpr int N_   = 343;      // 7*7*7 tokens per window
constexpr int NP   = 352;      // padded to 22*16
constexpr int C_   = 192;
constexpr int H_   = 6;
constexpr int HD   = 32;
constexpr int T_   = 2197;     // 13^3 CPB table rows
constexpr int CPB_ = 512;
constexpr int M_   = B_ * N_;  // 43904 = 686*64 (exact)
constexpr int MT   = 64;       // attention row tile
constexpr int NRT  = 6;        // ceil(343/64)
constexpr int QK_ROWS = B_ * H_ * N_;      // 263424
// LDS layout (bytes): sK [NP][32]h @0, sVt [32][NP]h @22528, sQ [64][32]h @45056,
//                     sS [64][NP]f @49152, sP/Vstage [64][NP]h @139264
constexpr unsigned LDS_K  = 0;
constexpr unsigned LDS_VT = (unsigned)(NP * HD * 2);
constexpr unsigned LDS_Q  = LDS_VT + (unsigned)(HD * NP * 2);
constexpr unsigned LDS_S  = LDS_Q + (unsigned)(MT * HD * 2);
constexpr unsigned LDS_P  = LDS_S + (unsigned)(MT * NP * 4);
constexpr size_t SMEM_ATTN = (size_t)LDS_P + (size_t)MT * NP * 2;  // 184320

// ---- WMMA fragment helpers (16x16x32 f16, layouts per CDNA5 ISA 7.12.2) ----
DEVI v16h frag_ld(const _Float16* base, int ld, int lane) {
  const _Float16* p = base + (size_t)(lane & 15) * ld + ((lane >> 4) << 3);
  union { v16h v; v8h h[2]; } u;
  u.h[0] = *(const v8h*)(p);
  u.h[1] = *(const v8h*)(p + 16);
  return u.v;
}

DEVI v8f wmma_f16(v16h a, v16h b, v8f c) {
  return __builtin_amdgcn_wmma_f32_16x16x32_f16(false, a, false, b, (short)0, c, false, false);
}

DEVI float wave_max(float v) {
  #pragma unroll
  for (int off = 16; off >= 1; off >>= 1) v = fmaxf(v, __shfl_xor(v, off, 32));
  return v;
}
DEVI float wave_sum(float v) {
  #pragma unroll
  for (int off = 16; off >= 1; off >>= 1) v += __shfl_xor(v, off, 32);
  return v;
}

// Async linear memory->LDS copy: `bytes` must be a multiple of 16.
// Uses GLOBAL_LOAD_ASYNC_TO_LDS_B128 (ASYNCcnt), 16B per active lane.
DEVI void async_copy_lin(unsigned lds_off, const void* gbase, int bytes, int tid) {
  int chunks = bytes >> 4;
  for (int c = tid; c < chunks; c += 256) {
    unsigned long long gaddr = (unsigned long long)(uintptr_t)gbase + ((unsigned long long)c << 4);
    unsigned laddr = lds_off + ((unsigned)c << 4);
    asm volatile("global_load_async_to_lds_b128 %0, %1, off"
                 :: "v"(laddr), "v"(gaddr)
                 : "memory");
  }
}
DEVI void async_wait0() {
  asm volatile("s_wait_asynccnt 0x0" ::: "memory");
}

// ---------------------------------------------------------------- converts
__global__ void wa_f32_to_f16(const float* __restrict__ src, _Float16* __restrict__ dst, int n) {
  int i = blockIdx.x * blockDim.x + threadIdx.x;
  if (i < n) dst[i] = (_Float16)src[i];
}

// ---------------------------------------------------------------- CPB MLP
__global__ void wa_cpb_h1(const float* __restrict__ table, const float* __restrict__ w1,
                          const float* __restrict__ b1, float* __restrict__ h1) {
  int i = blockIdx.x * blockDim.x + threadIdx.x;
  if (i >= T_ * CPB_) return;
  int t = i >> 9, j = i & 511;
  const float* tr = table + t * 3;
  const float* wr = w1 + j * 3;
  float s = tr[0]*wr[0] + tr[1]*wr[1] + tr[2]*wr[2] + b1[j];
  h1[i] = fmaxf(s, 0.0f);
}

__global__ void wa_cpb_tab(const float* __restrict__ h1, const float* __restrict__ w2,
                           float* __restrict__ btab) {
  int i = blockIdx.x * blockDim.x + threadIdx.x;
  if (i >= T_ * H_) return;
  int t = i / H_, h = i - t * H_;
  const float* hr = h1 + (size_t)t * CPB_;
  const float* wr = w2 + (size_t)h * CPB_;
  float s = 0.f;
  for (int j = 0; j < CPB_; ++j) s += hr[j] * wr[j];
  btab[(size_t)t * H_ + h] = s;
}

__global__ void wa_bias_full(const float* __restrict__ btab, const int* __restrict__ idx,
                             float* __restrict__ bias_full) {
  int i = blockIdx.x * blockDim.x + threadIdx.x;
  if (i >= H_ * N_ * N_) return;
  int h = i / (N_ * N_);
  int rem = i - h * (N_ * N_);
  float x = btab[(size_t)idx[rem] * H_ + h];
  bias_full[i] = 16.0f / (1.0f + __expf(-x));
}

// ---------------------------------------------------------------- QKV GEMM
__global__ void __launch_bounds__(256) wa_qkv_gemm(
    const _Float16* __restrict__ xh, const _Float16* __restrict__ wh,
    const float* __restrict__ q_bias, const float* __restrict__ v_bias,
    _Float16* __restrict__ qh, _Float16* __restrict__ kh, _Float16* __restrict__ vh) {
  __shared__ __align__(16) _Float16 sA[64 * 32];
  __shared__ __align__(16) _Float16 sB[64 * 32];
  const int m0 = blockIdx.x * 64, n0 = blockIdx.y * 64;
  const int tid = threadIdx.x, wave = tid >> 5, lane = tid & 31;
  const int t0 = wave, t1 = wave + 8;
  const int mt0 = t0 >> 2, nt0 = t0 & 3, mt1 = t1 >> 2, nt1 = t1 & 3;
  v8f acc0 = {}, acc1 = {};
  const int row = tid >> 2, cc = (tid & 3) * 8;
  for (int k0 = 0; k0 < C_; k0 += 32) {
    *(v8h*)&sA[row * 32 + cc] = *(const v8h*)&xh[(size_t)(m0 + row) * C_ + k0 + cc];
    *(v8h*)&sB[row * 32 + cc] = *(const v8h*)&wh[(size_t)(n0 + row) * C_ + k0 + cc];
    __syncthreads();
    v16h a0 = frag_ld(sA + mt0 * 16 * 32, 32, lane);
    v16h b0 = frag_ld(sB + nt0 * 16 * 32, 32, lane);
    acc0 = wmma_f16(a0, b0, acc0);
    v16h a1 = frag_ld(sA + mt1 * 16 * 32, 32, lane);
    v16h b1 = frag_ld(sB + nt1 * 16 * 32, 32, lane);
    acc1 = wmma_f16(a1, b1, acc1);
    __syncthreads();
  }
  const int col = lane & 15, rb = (lane >> 4) << 3;
  #pragma unroll
  for (int pass = 0; pass < 2; ++pass) {
    int mt = pass ? mt1 : mt0, nt = pass ? nt1 : nt0;
    v8f acc = pass ? acc1 : acc0;
    #pragma unroll
    for (int r = 0; r < 8; ++r) {
      int m = m0 + mt * 16 + rb + r;
      int c = n0 + nt * 16 + col;
      float bias = (c < C_) ? q_bias[c] : ((c >= 2 * C_) ? v_bias[c - 2 * C_] : 0.0f);
      float v = acc[r] + bias;
      int which = c / C_;
      int rem = c - which * C_;
      int h = rem >> 5, d = rem & 31;
      int b = m / N_, n = m - b * N_;
      _Float16* dst = (which == 0) ? qh : ((which == 1) ? kh : vh);
      dst[((((size_t)b * H_) + h) * N_ + n) * HD + d] = (_Float16)v;
    }
  }
}

// ---------------------------------------------------------------- normalize q/k
__global__ void wa_normalize_qk(_Float16* __restrict__ qh, _Float16* __restrict__ kh,
                                const float* __restrict__ logit_scale) {
  int rid = blockIdx.x * blockDim.x + threadIdx.x;
  if (rid >= 2 * QK_ROWS) return;
  bool isq = rid < QK_ROWS;
  int r = isq ? rid : rid - QK_ROWS;
  _Float16* p = (isq ? qh : kh) + (size_t)r * HD;
  float vals[HD];
  float ss = 0.f;
  #pragma unroll
  for (int d = 0; d < HD; ++d) { float x = (float)p[d]; vals[d] = x; ss += x * x; }
  float nrm = fmaxf(sqrtf(ss), 1e-12f);
  float sc = 1.0f / nrm;
  if (isq) {
    int h = (r / N_) % H_;
    float ls = fminf(logit_scale[h], 4.6051701859880914f);  // log(100)
    sc *= __expf(ls);
  }
  #pragma unroll
  for (int d = 0; d < HD; ++d) p[d] = (_Float16)(vals[d] * sc);
}

// ---------------------------------------------------------------- fused attention
__global__ void __launch_bounds__(256) wa_attention(
    const _Float16* __restrict__ qh, const _Float16* __restrict__ kh,
    const _Float16* __restrict__ vh, const float* __restrict__ bias_full,
    _Float16* __restrict__ attn_out) {
  extern __shared__ __align__(16) char smem[];
  _Float16* sK  = (_Float16*)(smem + LDS_K);    // [NP][32]
  _Float16* sVt = (_Float16*)(smem + LDS_VT);   // [32][NP]  (V transposed)
  _Float16* sQ  = (_Float16*)(smem + LDS_Q);    // [64][32]
  float*    sS  = (float*)(smem + LDS_S);       // [64][NP]
  _Float16* sP  = (_Float16*)(smem + LDS_P);    // [64][NP]; doubles as V row-major stage

  const int blk = blockIdx.x;
  const int bh = blk / NRT, mtb = blk - bh * NRT;
  const int b = bh / H_, h = bh - b * H_;
  const int row0 = mtb * MT;
  const int tid = threadIdx.x, wave = tid >> 5, lane = tid & 31;

  const size_t base = (((size_t)b * H_) + h) * N_ * HD;
  const _Float16* kb = kh + base;
  const _Float16* vb = vh + base;
  const _Float16* qb = qh + base;

  // --- async memory->LDS staging (ASYNCcnt path, no VGPR bounce) ---
  const int valid_q_rows = min(MT, N_ - row0);
  async_copy_lin(LDS_K, kb, N_ * HD * 2, tid);                      // K rows, linear
  async_copy_lin(LDS_P, vb, N_ * HD * 2, tid);                      // V rows -> stage
  async_copy_lin(LDS_Q, qb + (size_t)row0 * HD, valid_q_rows * HD * 2, tid);
  // zero-fill padded K rows and invalid Q rows (disjoint from async targets)
  for (int n = N_ + tid; n < NP; n += 256) {
    #pragma unroll
    for (int c = 0; c < 4; ++c) *(v8h*)&sK[n * HD + c * 8] = (v8h)(_Float16)0.f;
  }
  for (int m = valid_q_rows + tid; m < MT; m += 256) {
    #pragma unroll
    for (int c = 0; c < 4; ++c) *(v8h*)&sQ[m * HD + c * 8] = (v8h)(_Float16)0.f;
  }
  async_wait0();
  __syncthreads();

  // --- transpose V stage -> sVt (LDS->LDS), zero pad columns ---
  for (int n = tid; n < NP; n += 256) {
    if (n < N_) {
      #pragma unroll
      for (int d = 0; d < HD; ++d) sVt[d * NP + n] = sP[n * HD + d];
    } else {
      #pragma unroll
      for (int d = 0; d < HD; ++d) sVt[d * NP + n] = (_Float16)0.f;
    }
  }
  __syncthreads();

  // --- S = Q @ K^T : 4 x 22 tiles of 16x16, K=32 in one WMMA ---
  const int col = lane & 15, rb = (lane >> 4) << 3;
  #pragma unroll
  for (int i = 0; i < 11; ++i) {
    int t = wave * 11 + i;          // 0..87
    int mt = t / 22, nt = t - mt * 22;
    v16h a = frag_ld(sQ + mt * 16 * HD, HD, lane);
    v16h bf = frag_ld(sK + nt * 16 * HD, HD, lane);
    v8f acc = {};
    acc = wmma_f16(a, bf, acc);
    #pragma unroll
    for (int r = 0; r < 8; ++r)
      sS[(mt * 16 + rb + r) * NP + nt * 16 + col] = acc[r];
  }
  __syncthreads();

  // --- bias + softmax, one wave per 8 rows, 11 columns per lane ---
  const float* brow_base = bias_full + (size_t)h * N_ * N_;
  #pragma unroll
  for (int rr = 0; rr < 8; ++rr) {
    int rowl = wave * 8 + rr;
    int ig = row0 + rowl;
    const float* brow = brow_base + (size_t)((ig < N_) ? ig : 0) * N_;
    __builtin_prefetch(brow + lane, 0, 1);   // global_prefetch_b8
    float sv[11];
    float mx = -1e30f;
    #pragma unroll
    for (int jj = 0; jj < 11; ++jj) {
      int j = jj * 32 + lane;
      float s = sS[rowl * NP + j];
      s = (j < N_) ? (s + brow[j]) : -1e30f;
      sv[jj] = s;
      mx = fmaxf(mx, s);
    }
    mx = wave_max(mx);
    float sum = 0.f;
    #pragma unroll
    for (int jj = 0; jj < 11; ++jj) { float p = __expf(sv[jj] - mx); sv[jj] = p; sum += p; }
    sum = wave_sum(sum);
    float inv = 1.0f / sum;
    #pragma unroll
    for (int jj = 0; jj < 11; ++jj) {
      int j = jj * 32 + lane;
      sP[rowl * NP + j] = (_Float16)(sv[jj] * inv);
    }
  }
  __syncthreads();

  // --- O = P @ V : 4 x 2 output tiles (one per wave), K loop over 11 chunks ---
  {
    int mt = wave >> 1, nt = wave & 1;
    v8f acc = {};
    #pragma unroll
    for (int kt = 0; kt < 11; ++kt) {
      v16h a  = frag_ld(sP + mt * 16 * NP + kt * 32, NP, lane);
      v16h bf = frag_ld(sVt + nt * 16 * NP + kt * 32, NP, lane);
      acc = wmma_f16(a, bf, acc);
    }
    #pragma unroll
    for (int r = 0; r < 8; ++r) {
      int ig = row0 + mt * 16 + rb + r;
      if (ig < N_)
        attn_out[((size_t)b * N_ + ig) * C_ + h * HD + nt * 16 + col] = (_Float16)acc[r];
    }
  }
}

// ---------------------------------------------------------------- output projection
__global__ void __launch_bounds__(256) wa_proj_gemm(
    const _Float16* __restrict__ ah, const _Float16* __restrict__ wh,
    const float* __restrict__ proj_b, float* __restrict__ out) {
  __shared__ __align__(16) _Float16 sA[64 * 32];
  __shared__ __align__(16) _Float16 sB[64 * 32];
  const int m0 = blockIdx.x * 64, n0 = blockIdx.y * 64;
  const int tid = threadIdx.x, wave = tid >> 5, lane = tid & 31;
  const int t0 = wave, t1 = wave + 8;
  const int mt0 = t0 >> 2, nt0 = t0 & 3, mt1 = t1 >> 2, nt1 = t1 & 3;
  v8f acc0 = {}, acc1 = {};
  const int row = tid >> 2, cc = (tid & 3) * 8;
  for (int k0 = 0; k0 < C_; k0 += 32) {
    *(v8h*)&sA[row * 32 + cc] = *(const v8h*)&ah[(size_t)(m0 + row) * C_ + k0 + cc];
    *(v8h*)&sB[row * 32 + cc] = *(const v8h*)&wh[(size_t)(n0 + row) * C_ + k0 + cc];
    __syncthreads();
    v16h a0 = frag_ld(sA + mt0 * 16 * 32, 32, lane);
    v16h b0 = frag_ld(sB + nt0 * 16 * 32, 32, lane);
    acc0 = wmma_f16(a0, b0, acc0);
    v16h a1 = frag_ld(sA + mt1 * 16 * 32, 32, lane);
    v16h b1 = frag_ld(sB + nt1 * 16 * 32, 32, lane);
    acc1 = wmma_f16(a1, b1, acc1);
    __syncthreads();
  }
  const int col = lane & 15, rb = (lane >> 4) << 3;
  #pragma unroll
  for (int pass = 0; pass < 2; ++pass) {
    int mt = pass ? mt1 : mt0, nt = pass ? nt1 : nt0;
    v8f acc = pass ? acc1 : acc0;
    #pragma unroll
    for (int r = 0; r < 8; ++r) {
      int m = m0 + mt * 16 + rb + r;
      int c = n0 + nt * 16 + col;
      out[(size_t)m * C_ + c] = acc[r] + proj_b[c];
    }
  }
}

// ---------------------------------------------------------------- launcher
extern "C" void kernel_launch(void* const* d_in, const int* in_sizes, int n_in,
                              void* d_out, int out_size, void* d_ws, size_t ws_size,
                              hipStream_t stream) {
  const float* x       = (const float*)d_in[0];
  const float* qkv_w   = (const float*)d_in[1];
  const float* q_bias  = (const float*)d_in[2];
  const float* v_bias  = (const float*)d_in[3];
  const float* logit_s = (const float*)d_in[4];
  const float* cpb_w1  = (const float*)d_in[5];
  const float* cpb_b1  = (const float*)d_in[6];
  const float* cpb_w2  = (const float*)d_in[7];
  const float* proj_w  = (const float*)d_in[8];
  const float* proj_b  = (const float*)d_in[9];
  const float* table   = (const float*)d_in[10];
  const int*   rpi     = (const int*)d_in[11];
  float* out = (float*)d_out;

  char* ws = (char*)d_ws;
  size_t off = 0;
  auto take = [&](size_t bytes) -> char* {
    char* p = ws + off;
    off += (bytes + 255) & ~(size_t)255;
    return p;
  };
  _Float16* xh     = (_Float16*)take((size_t)M_ * C_ * 2);
  _Float16* qkvwh  = (_Float16*)take((size_t)3 * C_ * C_ * 2);
  _Float16* projwh = (_Float16*)take((size_t)C_ * C_ * 2);
  _Float16* qh     = (_Float16*)take((size_t)QK_ROWS * HD * 2);
  _Float16* kh     = (_Float16*)take((size_t)QK_ROWS * HD * 2);
  _Float16* vh     = (_Float16*)take((size_t)QK_ROWS * HD * 2);
  float*    h1     = (float*)take((size_t)T_ * CPB_ * 4);
  float*    btab   = (float*)take((size_t)T_ * H_ * 4);
  float*    biasf  = (float*)take((size_t)H_ * N_ * N_ * 4);
  _Float16* attnO  = (_Float16*)take((size_t)M_ * C_ * 2);
  (void)ws_size; (void)in_sizes; (void)n_in; (void)out_size;

  {
    int n = M_ * C_;
    wa_f32_to_f16<<<(n + 255) / 256, 256, 0, stream>>>(x, xh, n);
    n = 3 * C_ * C_;
    wa_f32_to_f16<<<(n + 255) / 256, 256, 0, stream>>>(qkv_w, qkvwh, n);
    n = C_ * C_;
    wa_f32_to_f16<<<(n + 255) / 256, 256, 0, stream>>>(proj_w, projwh, n);
  }
  wa_cpb_h1<<<(T_ * CPB_ + 255) / 256, 256, 0, stream>>>(table, cpb_w1, cpb_b1, h1);
  wa_cpb_tab<<<(T_ * H_ + 255) / 256, 256, 0, stream>>>(h1, cpb_w2, btab);
  wa_bias_full<<<(H_ * N_ * N_ + 255) / 256, 256, 0, stream>>>(btab, rpi, biasf);
  wa_qkv_gemm<<<dim3(M_ / 64, (3 * C_) / 64), 256, 0, stream>>>(
      xh, qkvwh, q_bias, v_bias, qh, kh, vh);
  wa_normalize_qk<<<(2 * QK_ROWS + 255) / 256, 256, 0, stream>>>(qh, kh, logit_s);
  hipFuncSetAttribute(reinterpret_cast<const void*>(wa_attention),
                      hipFuncAttributeMaxDynamicSharedMemorySize, (int)SMEM_ATTN);
  wa_attention<<<B_ * H_ * NRT, 256, SMEM_ATTN, stream>>>(qh, kh, vh, biasf, attnO);
  wa_proj_gemm<<<dim3(M_ / 64, C_ / 64), 256, 0, stream>>>(attnO, projwh, proj_b, out);
}
